// BCRNNlayer_7971459301876
// MI455X (gfx1250) — matrix operand order repaired and checked by reading, not compiled, not run on hardware
//
#include <hip/hip_runtime.h>
#include <cstdint>

typedef __attribute__((ext_vector_type(16))) _Float16 v16h;
typedef __attribute__((ext_vector_type(8)))  float    v8f;

#define NT   12
#define NXY  160
#define NPIX (NXY*NXY)     // 25600
#define HID  64
#define TILEW 16           // pixels (one row segment) per workgroup
#define SROW  18           // staged cols (16 + halo)
#define SSTR  72           // LDS pixel stride in halfs (144B: 16B aligned, bank-spread)
#define NTILES (NPIX / TILEW)   // 1600

__device__ __forceinline__ int reflect_idx(int i, int n) {
    i = (i < 0) ? -i : i;
    return (i >= n) ? (2 * n - 2 - i) : i;
}

union BFrag { uint32_t u[8]; v16h h; };
union AccU  { v8f v; float f[8]; };
union PkU   { _Float16 h2[2]; uint32_t u; };

// ---------------------------------------------------------------------------
// Pack f32 OIHW weights into WMMA A-fragment layout (16-bit A 16x32, ISA 7.12.2)
// K ordering: k = tap*C + c  (tap-major). dst dword index:
//   (((q*4 + mtile)*32 + lane)*8 + j)
// ---------------------------------------------------------------------------
__global__ void pack_w(const float* __restrict__ w, uint32_t* __restrict__ dst,
                       int Cin, int Ktot, int total) {
    int tid = blockIdx.x * blockDim.x + threadIdx.x;
    if (tid >= total) return;
    int j    = tid & 7;
    int lane = (tid >> 3) & 31;
    int mt   = (tid >> 8) & 3;
    int q    = tid >> 10;
    int m  = lane & 15;
    int hh = lane >> 4;
    int K0 = 16 * hh + 2 * j;
    int o  = mt * 16 + m;
    PkU pk;
#pragma unroll
    for (int e = 0; e < 2; ++e) {
        int k = q * 32 + K0 + e;
        float val = 0.f;
        if (k < Ktot) {
            int tap = k / Cin;
            int c   = k - tap * Cin;
            val = w[((size_t)o * Cin + c) * 9 + tap];
        }
        pk.h2[e] = (_Float16)val;
    }
    dst[tid] = pk.u;
}

// ---------------------------------------------------------------------------
// Transpose planar f32 [C][NT][NPIX] -> channel-last f16 [NT*NPIX][C]
// ---------------------------------------------------------------------------
__global__ void trans_cl64(const float* __restrict__ src, _Float16* __restrict__ dst) {
    int idx = blockIdx.x * blockDim.x + threadIdx.x;
    if (idx >= NT * NPIX) return;
    int t = idx / NPIX, pix = idx - t * NPIX;
    _Float16 buf[64] __attribute__((aligned(16)));
#pragma unroll
    for (int c = 0; c < 64; ++c)
        buf[c] = (_Float16)src[((size_t)c * NT + t) * NPIX + pix];
    uint4* d = (uint4*)&dst[(size_t)idx * 64];
    const uint4* s = (const uint4*)buf;
#pragma unroll
    for (int k = 0; k < 8; ++k) d[k] = s[k];
}

__global__ void trans_cl2(const float* __restrict__ src, _Float16* __restrict__ dst) {
    int idx = blockIdx.x * blockDim.x + threadIdx.x;
    if (idx >= NT * NPIX) return;
    int t = idx / NPIX, pix = idx - t * NPIX;
    PkU pk;
    pk.h2[0] = (_Float16)src[((size_t)0 * NT + t) * NPIX + pix];
    pk.h2[1] = (_Float16)src[((size_t)1 * NT + t) * NPIX + pix];
    *(uint32_t*)&dst[(size_t)idx * 2] = pk.u;
}

// ---------------------------------------------------------------------------
// pre[t][o][pix] = cconv(iter, w_ih) + cconv(input, w_i2h) + b_ih + b_i2h
// grid: (NTILES, NT), block: 128 (4 waves, one 16-out-channel tile each)
// ---------------------------------------------------------------------------
__global__ __launch_bounds__(128) void bcrnn_pre(
    const _Float16* __restrict__ itR, const _Float16* __restrict__ itI,
    const _Float16* __restrict__ inR, const _Float16* __restrict__ inI,
    const uint32_t* __restrict__ wihR, const uint32_t* __restrict__ wihI,
    const uint32_t* __restrict__ wi2R, const uint32_t* __restrict__ wi2I,
    const float* __restrict__ bihR, const float* __restrict__ bihI,
    const float* __restrict__ bi2R, const float* __restrict__ bi2I,
    float* __restrict__ preR, float* __restrict__ preI) {

    __shared__ _Float16 sR[3 * SROW * SSTR] __attribute__((aligned(16)));
    __shared__ _Float16 sI[3 * SROW * SSTR] __attribute__((aligned(16)));
    __shared__ _Float16 s2R[3 * SROW * 2] __attribute__((aligned(16)));
    __shared__ _Float16 s2I[3 * SROW * 2] __attribute__((aligned(16)));

    const int t    = blockIdx.y;
    const int row  = blockIdx.x / (NXY / TILEW);
    const int col0 = (blockIdx.x % (NXY / TILEW)) * TILEW;
    const int tid  = threadIdx.x;

    const _Float16* itRb = itR + (size_t)t * NPIX * 64;
    const _Float16* itIb = itI + (size_t)t * NPIX * 64;

    for (int item = tid; item < 3 * SROW * 8; item += 128) {
        int pix = item >> 3, chunk = item & 7;
        int r = pix / SROW, cc = pix - r * SROW;
        int gr = reflect_idx(row + r - 1, NXY);
        int gc = reflect_idx(col0 + cc - 1, NXY);
        size_t goff = ((size_t)(gr * NXY + gc)) * 64 + chunk * 8;
        *(uint4*)&sR[pix * SSTR + chunk * 8] = *(const uint4*)&itRb[goff];
        *(uint4*)&sI[pix * SSTR + chunk * 8] = *(const uint4*)&itIb[goff];
    }
    if (tid < 3 * SROW) {
        int r = tid / SROW, cc = tid - r * SROW;
        int gr = reflect_idx(row + r - 1, NXY);
        int gc = reflect_idx(col0 + cc - 1, NXY);
        size_t goff = ((size_t)t * NPIX + (size_t)(gr * NXY + gc)) * 2;
        *(uint32_t*)&s2R[tid * 2] = *(const uint32_t*)&inR[goff];
        *(uint32_t*)&s2I[tid * 2] = *(const uint32_t*)&inI[goff];
    }
    __syncthreads();

    const int wave = tid >> 5, lane = tid & 31;
    const int p = lane & 15, hh = lane >> 4;

    v8f accP = {}, accQ = {}, accS = {};

    for (int q = 0; q < 18; ++q) {
        int tap = q >> 1, cb = (q & 1) * 32;
        int dy = tap / 3 - 1, dx = tap % 3 - 1;
        int pixb = (1 + dy) * SROW + (1 + p + dx);
        BFrag bR, bI;
        int cbase = cb + 16 * hh;
#pragma unroll
        for (int j = 0; j < 8; ++j) {
            int c = cbase + 2 * j;
            bR.u[j] = *(const uint32_t*)&sR[pixb * SSTR + c];
            bI.u[j] = *(const uint32_t*)&sI[pixb * SSTR + c];
        }
        const v16h aR = *(const v16h*)&wihR[(((q * 4 + wave) * 32) + lane) * 8];
        const v16h aI = *(const v16h*)&wihI[(((q * 4 + wave) * 32) + lane) * 8];
        accP = __builtin_amdgcn_wmma_f32_16x16x32_f16(false, aR, false, bR.h, (short)0, accP, false, false);
        accQ = __builtin_amdgcn_wmma_f32_16x16x32_f16(false, aI, false, bI.h, (short)0, accQ, false, false);
        accS = __builtin_amdgcn_wmma_f32_16x16x32_f16(false, aR, false, bI.h, (short)0, accS, false, false);
        accS = __builtin_amdgcn_wmma_f32_16x16x32_f16(false, aI, false, bR.h, (short)0, accS, false, false);
    }

    // input-to-hidden chunk: K = tap*2 + c (18 valid of 32, A zero-padded)
    {
        BFrag bR, bI;
#pragma unroll
        for (int j = 0; j < 8; ++j) {
            int klocal = 16 * hh + 2 * j;
            bool valid = (klocal <= 16);
            int tapc = valid ? (klocal >> 1) : 0;
            int dy = tapc / 3 - 1, dx = tapc % 3 - 1;
            int pixb = (1 + dy) * SROW + (1 + p + dx);
            uint32_t vR = *(const uint32_t*)&s2R[pixb * 2];
            uint32_t vI = *(const uint32_t*)&s2I[pixb * 2];
            bR.u[j] = valid ? vR : 0u;
            bI.u[j] = valid ? vI : 0u;
        }
        const v16h aR = *(const v16h*)&wi2R[(wave * 32 + lane) * 8];
        const v16h aI = *(const v16h*)&wi2I[(wave * 32 + lane) * 8];
        accP = __builtin_amdgcn_wmma_f32_16x16x32_f16(false, aR, false, bR.h, (short)0, accP, false, false);
        accQ = __builtin_amdgcn_wmma_f32_16x16x32_f16(false, aI, false, bI.h, (short)0, accQ, false, false);
        accS = __builtin_amdgcn_wmma_f32_16x16x32_f16(false, aR, false, bI.h, (short)0, accS, false, false);
        accS = __builtin_amdgcn_wmma_f32_16x16x32_f16(false, aI, false, bR.h, (short)0, accS, false, false);
    }

    AccU uP, uQ, uS; uP.v = accP; uQ.v = accQ; uS.v = accS;
    int x = col0 + p;
    size_t pix = (size_t)row * NXY + x;
#pragma unroll
    for (int v = 0; v < 8; ++v) {
        int o = wave * 16 + v + 8 * hh;
        size_t pidx = ((size_t)(t * 64 + o)) * NPIX + pix;
        preR[pidx] = uP.f[v] - uQ.f[v] + bihR[o] + bi2R[o];
        preI[pidx] = uS.f[v] + bihI[o] + bi2I[o];
    }
}

// ---------------------------------------------------------------------------
// one RNN step: z = pre[t] + cconv(h_prev, w_h2h) + b_h2h; h = modrelu(z)
// dir==0: out = h (write)   dir==1: out += h (accumulate)
// ---------------------------------------------------------------------------
__global__ __launch_bounds__(128) void bcrnn_step(
    const _Float16* __restrict__ hpR, const _Float16* __restrict__ hpI,
    _Float16* __restrict__ hnR, _Float16* __restrict__ hnI,
    const float* __restrict__ preR, const float* __restrict__ preI,
    const uint32_t* __restrict__ wpR, const uint32_t* __restrict__ wpI,
    const float* __restrict__ bR, const float* __restrict__ bI,
    const float* __restrict__ modb, float* __restrict__ out,
    int t, int dir) {

    __shared__ _Float16 sR[3 * SROW * SSTR] __attribute__((aligned(16)));
    __shared__ _Float16 sI[3 * SROW * SSTR] __attribute__((aligned(16)));

    const int row  = blockIdx.x / (NXY / TILEW);
    const int col0 = (blockIdx.x % (NXY / TILEW)) * TILEW;
    const int tid  = threadIdx.x;

    for (int item = tid; item < 3 * SROW * 8; item += 128) {
        int pix = item >> 3, chunk = item & 7;
        int r = pix / SROW, cc = pix - r * SROW;
        int gr = reflect_idx(row + r - 1, NXY);
        int gc = reflect_idx(col0 + cc - 1, NXY);
        size_t goff = ((size_t)(gr * NXY + gc)) * 64 + chunk * 8;
        *(uint4*)&sR[pix * SSTR + chunk * 8] = *(const uint4*)&hpR[goff];
        *(uint4*)&sI[pix * SSTR + chunk * 8] = *(const uint4*)&hpI[goff];
    }
    __syncthreads();

    const int wave = tid >> 5, lane = tid & 31;
    const int p = lane & 15, hh = lane >> 4;

    v8f accP = {}, accQ = {}, accS = {};

    for (int q = 0; q < 18; ++q) {
        int tap = q >> 1, cb = (q & 1) * 32;
        int dy = tap / 3 - 1, dx = tap % 3 - 1;
        int pixb = (1 + dy) * SROW + (1 + p + dx);
        BFrag bRf, bIf;
        int cbase = cb + 16 * hh;
#pragma unroll
        for (int j = 0; j < 8; ++j) {
            int c = cbase + 2 * j;
            bRf.u[j] = *(const uint32_t*)&sR[pixb * SSTR + c];
            bIf.u[j] = *(const uint32_t*)&sI[pixb * SSTR + c];
        }
        const v16h aR = *(const v16h*)&wpR[(((q * 4 + wave) * 32) + lane) * 8];
        const v16h aI = *(const v16h*)&wpI[(((q * 4 + wave) * 32) + lane) * 8];
        accP = __builtin_amdgcn_wmma_f32_16x16x32_f16(false, aR, false, bRf.h, (short)0, accP, false, false);
        accQ = __builtin_amdgcn_wmma_f32_16x16x32_f16(false, aI, false, bIf.h, (short)0, accQ, false, false);
        accS = __builtin_amdgcn_wmma_f32_16x16x32_f16(false, aR, false, bIf.h, (short)0, accS, false, false);
        accS = __builtin_amdgcn_wmma_f32_16x16x32_f16(false, aI, false, bRf.h, (short)0, accS, false, false);
    }

    AccU uP, uQ, uS; uP.v = accP; uQ.v = accQ; uS.v = accS;
    int x = col0 + p;
    size_t pix = (size_t)row * NXY + x;
#pragma unroll
    for (int v = 0; v < 8; ++v) {
        int o = wave * 16 + v + 8 * hh;
        size_t pidx = ((size_t)(t * 64 + o)) * NPIX + pix;
        float zr = uP.f[v] - uQ.f[v] + preR[pidx] + bR[o];
        float zi = uS.f[v] + preI[pidx] + bI[o];
        float mag = sqrtf(zr * zr + zi * zi);
        float mb = mag + modb[o];
        float scale = (mb > 0.f ? mb : 0.f) / (mag + 1e-8f);
        float hr = scale * zr, hi = scale * zi;
        hnR[pix * 64 + o] = (_Float16)hr;
        hnI[pix * 64 + o] = (_Float16)hi;
        size_t obase = (((size_t)o * NT + t) * NPIX + pix) * 2;
        if (dir == 0) { out[obase] = hr; out[obase + 1] = hi; }
        else          { out[obase] += hr; out[obase + 1] += hi; }
    }
}

// ---------------------------------------------------------------------------
extern "C" void kernel_launch(void* const* d_in, const int* in_sizes, int n_in,
                              void* d_out, int out_size, void* d_ws, size_t ws_size,
                              hipStream_t stream) {
    const float* input_r = (const float*)d_in[0];
    const float* input_i = (const float*)d_in[1];
    const float* iter_r  = (const float*)d_in[2];
    const float* iter_i  = (const float*)d_in[3];
    const float* w_i2h_r = (const float*)d_in[4];
    const float* w_i2h_i = (const float*)d_in[5];
    const float* b_i2h_r = (const float*)d_in[6];
    const float* b_i2h_i = (const float*)d_in[7];
    const float* w_h2h_r = (const float*)d_in[8];
    const float* w_h2h_i = (const float*)d_in[9];
    const float* b_h2h_r = (const float*)d_in[10];
    const float* b_h2h_i = (const float*)d_in[11];
    const float* w_ih_r  = (const float*)d_in[12];
    const float* w_ih_i  = (const float*)d_in[13];
    const float* b_ih_r  = (const float*)d_in[14];
    const float* b_ih_i  = (const float*)d_in[15];
    const float* mod_b   = (const float*)d_in[16];
    float* out = (float*)d_out;

    char* ws = (char*)d_ws;
    const size_t S_PRE = (size_t)NT * 64 * NPIX * 4;   // 78,643,200
    const size_t S_IT  = (size_t)NT * NPIX * 64 * 2;   // 39,321,600
    const size_t S_IN  = (size_t)NT * NPIX * 2 * 2;    //  1,228,800
    const size_t S_H   = (size_t)NPIX * 64 * 2;        //  3,276,800
    const size_t S_WB  = (size_t)18 * 1024 * 4;        //     73,728
    const size_t S_WS  = (size_t)1024 * 4;             //      4,096

    size_t off = 0;
    float*     preR = (float*)(ws + off);      off += S_PRE;
    float*     preI = (float*)(ws + off);      off += S_PRE;
    _Float16*  itR  = (_Float16*)(ws + off);   off += S_IT;
    _Float16*  itI  = (_Float16*)(ws + off);   off += S_IT;
    _Float16*  inR  = (_Float16*)(ws + off);   off += S_IN;
    _Float16*  inI  = (_Float16*)(ws + off);   off += S_IN;
    _Float16*  h0R  = (_Float16*)(ws + off);   off += S_H;
    _Float16*  h0I  = (_Float16*)(ws + off);   off += S_H;
    _Float16*  h1R  = (_Float16*)(ws + off);   off += S_H;
    _Float16*  h1I  = (_Float16*)(ws + off);   off += S_H;
    uint32_t*  wpHR = (uint32_t*)(ws + off);   off += S_WB;
    uint32_t*  wpHI = (uint32_t*)(ws + off);   off += S_WB;
    uint32_t*  wpGR = (uint32_t*)(ws + off);   off += S_WB;   // w_ih
    uint32_t*  wpGI = (uint32_t*)(ws + off);   off += S_WB;
    uint32_t*  wpXR = (uint32_t*)(ws + off);   off += S_WS;   // w_i2h
    uint32_t*  wpXI = (uint32_t*)(ws + off);   off += S_WS;

    // 1) pack weights into WMMA A-fragment layout
    pack_w<<<72, 256, 0, stream>>>(w_h2h_r, wpHR, 64, 576, 18432);
    pack_w<<<72, 256, 0, stream>>>(w_h2h_i, wpHI, 64, 576, 18432);
    pack_w<<<72, 256, 0, stream>>>(w_ih_r,  wpGR, 64, 576, 18432);
    pack_w<<<72, 256, 0, stream>>>(w_ih_i,  wpGI, 64, 576, 18432);
    pack_w<<<4, 256, 0, stream>>>(w_i2h_r, wpXR, 2, 18, 1024);
    pack_w<<<4, 256, 0, stream>>>(w_i2h_i, wpXI, 2, 18, 1024);

    // 2) activations -> channel-last f16
    trans_cl64<<<(NT * NPIX + 127) / 128, 128, 0, stream>>>(iter_r, itR);
    trans_cl64<<<(NT * NPIX + 127) / 128, 128, 0, stream>>>(iter_i, itI);
    trans_cl2 <<<(NT * NPIX + 127) / 128, 128, 0, stream>>>(input_r, inR);
    trans_cl2 <<<(NT * NPIX + 127) / 128, 128, 0, stream>>>(input_i, inI);

    // 3) pre = i2h(input) + ih(iter) + biases
    bcrnn_pre<<<dim3(NTILES, NT), 128, 0, stream>>>(
        itR, itI, inR, inI, wpGR, wpGI, wpXR, wpXI,
        b_ih_r, b_ih_i, b_i2h_r, b_i2h_i, preR, preI);

    _Float16* hbR[2] = {h0R, h1R};
    _Float16* hbI[2] = {h0I, h1I};

    // 4) forward scan (writes out)
    hipMemsetAsync(h0R, 0, S_H, stream);
    hipMemsetAsync(h0I, 0, S_H, stream);
    for (int t = 0; t < NT; ++t) {
        bcrnn_step<<<NTILES, 128, 0, stream>>>(
            hbR[t & 1], hbI[t & 1], hbR[(t + 1) & 1], hbI[(t + 1) & 1],
            preR, preI, wpHR, wpHI, b_h2h_r, b_h2h_i, mod_b, out, t, 0);
    }

    // 5) backward scan (accumulates into out)
    hipMemsetAsync(h0R, 0, S_H, stream);
    hipMemsetAsync(h0I, 0, S_H, stream);
    for (int k = 0; k < NT; ++k) {
        int t = NT - 1 - k;
        bcrnn_step<<<NTILES, 128, 0, stream>>>(
            hbR[k & 1], hbI[k & 1], hbR[(k + 1) & 1], hbI[(k + 1) & 1],
            preR, preI, wpHR, wpHI, b_h2h_r, b_h2h_i, mod_b, out, t, 1);
    }
}
